// MaskedScaledDotProductAttention_67585605370702
// MI455X (gfx1250) — matrix operand order repaired and checked by reading, not compiled, not run on hardware
//
#include <hip/hip_runtime.h>
#include <hip/hip_bf16.h>
#include <stdint.h>

// MI455X (gfx1250) flash-attention, causal, bf16 WMMA with fp32 softmax/accum.
// B=16, S=2048, D=128.  Compute-bound (~34 dense GFLOP vs ~67MB of HBM traffic),
// so everything funnels through v_wmma_f32_16x16x32_bf16.

typedef __attribute__((ext_vector_type(16))) __bf16 v16bf;
typedef __attribute__((ext_vector_type(8)))  __bf16 v8bf;
typedef __attribute__((ext_vector_type(8)))  float  v8f;

#define BATCH 16
#define SEQ   2048
#define DIM   128
#define BM    128        // query rows per workgroup (8 waves x 16 rows)
#define BN    32         // keys per KV iteration
#define KSTR  136        // LDS row stride (elements) for K tile  (272B, 16B-aligned, conflict-free)
#define VTSTR 40         // LDS row stride (elements) for V^T tile (80B, 16B-aligned, conflict-free)
#define PSTR  32         // LDS row stride for per-wave P scratch

static __device__ __forceinline__ __bf16 f2bf(float f) {
  // round-to-nearest-even fp32 -> bf16
  unsigned int u = __builtin_bit_cast(unsigned int, f);
  u += 0x7FFFu + ((u >> 16) & 1u);
  unsigned short h = (unsigned short)(u >> 16);
  return __builtin_bit_cast(__bf16, h);
}

// Load 16 consecutive bf16 from LDS (16B-aligned) as a WMMA B-operand half-row pair.
static __device__ __forceinline__ v16bf ld_b16x16(const __bf16* p) {
  v8bf a = *(const v8bf*)p;
  v8bf b = *(const v8bf*)(p + 8);
  v16bf r;
#pragma unroll
  for (int j = 0; j < 8; ++j) { r[j] = a[j]; r[8 + j] = b[j]; }
  return r;
}

__global__ __launch_bounds__(256)
void MaskedScaledDotProductAttention_67585605370702_kernel(
    const float* __restrict__ Q, const float* __restrict__ K,
    const float* __restrict__ V, float* __restrict__ O)
{
  __shared__ __align__(16) __bf16 sK [BN * KSTR];    // K tile, row-major [key][d]
  __shared__ __align__(16) __bf16 sVT[DIM * VTSTR];  // V tile, transposed [dv][key]
  __shared__ __align__(16) __bf16 sP [8 * 16 * PSTR];// per-wave P scratch (C-layout -> A-layout)

  const int tid   = threadIdx.x;
  const int wave  = tid >> 5;
  const int lane  = tid & 31;
  const int l16   = lane & 15;
  const bool hi   = lane >= 16;
  const int rowoff = hi ? 8 : 0;          // C-layout: VGPR v holds rows v (lo half) / v+8 (hi half)
  const int kb     = hi ? 8 : 0;          // half the K base of this lane half (K starts at 2kb)

  const int batch = blockIdx.y;
  const int q0    = blockIdx.x * BM;      // workgroup's first query row
  const int qw    = q0 + wave * 16;       // this wave's first query row
  const int qrow  = qw + l16;             // this lane's A-matrix row

  const float* Qb = Q + ((size_t)(batch * SEQ + qrow)) * DIM;
  const float* Kb = K + (size_t)batch * SEQ * DIM;
  const float* Vb = V + (size_t)batch * SEQ * DIM;
  float*       Ob = O + (size_t)batch * SEQ * DIM;

  // ---- Q -> bf16 A-operand registers (4 chunks of Kdim=32, contiguous K = 2kb + s per lane half) ----
  v16bf qa[4];
#pragma unroll
  for (int c = 0; c < 4; ++c) {
#pragma unroll
    for (int j = 0; j < 8; ++j) {
      qa[c][j]     = f2bf(Qb[c * 32 + 2 * kb + j]);        // K = 2kb .. 2kb+7
      qa[c][8 + j] = f2bf(Qb[c * 32 + 2 * kb + 8 + j]);    // K = 2kb+8 .. 2kb+15
    }
  }

  // ---- flash-attention state ----
  v8f oacc[8];
#pragma unroll
  for (int t = 0; t < 8; ++t)
#pragma unroll
    for (int v = 0; v < 8; ++v) oacc[t][v] = 0.0f;

  float mrow[8], lrow[8];
#pragma unroll
  for (int v = 0; v < 8; ++v) { mrow[v] = -3.0e38f; lrow[v] = 0.0f; }

  __bf16* Pmy = sP + wave * 16 * PSTR;
  const float scale = 0.08838834764831845f;   // 1/sqrt(128)
  const int nblocks = (q0 + BM) / BN;         // causal: only blocks up to the tile diagonal

  for (int nb = 0; nb < nblocks; ++nb) {
    const int kvbase = nb * BN;
    __syncthreads();

    // ---- cooperative LDS fill: K row-major bf16, V transposed bf16 ----
#pragma unroll
    for (int r = 0; r < 4; ++r) {
      const int idx4 = tid + 256 * r;              // 1024 float4 per tile
      const int key  = idx4 >> 5;                  // 0..31
      const int d    = (idx4 & 31) << 2;           // 0..124 step 4
      const float4 kf = *(const float4*)(Kb + (size_t)(kvbase + key) * DIM + d);
      const float4 vf = *(const float4*)(Vb + (size_t)(kvbase + key) * DIM + d);
      sK[key * KSTR + d + 0] = f2bf(kf.x);
      sK[key * KSTR + d + 1] = f2bf(kf.y);
      sK[key * KSTR + d + 2] = f2bf(kf.z);
      sK[key * KSTR + d + 3] = f2bf(kf.w);
      sVT[(d + 0) * VTSTR + key] = f2bf(vf.x);
      sVT[(d + 1) * VTSTR + key] = f2bf(vf.y);
      sVT[(d + 2) * VTSTR + key] = f2bf(vf.z);
      sVT[(d + 3) * VTSTR + key] = f2bf(vf.w);
    }
    // prefetch next KV block into cache (global_prefetch_b8)
    if (nb + 1 < nblocks) {
      __builtin_prefetch(Kb + (size_t)(kvbase + BN) * DIM + tid * 16, 0, 1);
      __builtin_prefetch(Vb + (size_t)(kvbase + BN) * DIM + tid * 16, 0, 1);
    }
    __syncthreads();

    // wave-uniform causal skip: keeps EXEC all-ones around WMMA
    if (kvbase > qw + 15) continue;
    const bool need_mask = (kvbase + BN - 1) > qw;

    // ---- S = Q K^T : two 16x16 tiles, Kdim = 128 via 4 WMMAs each ----
    v8f sacc[2];
#pragma unroll
    for (int ns = 0; ns < 2; ++ns) {
#pragma unroll
      for (int v = 0; v < 8; ++v) sacc[ns][v] = 0.0f;
#pragma unroll
      for (int c = 0; c < 4; ++c) {
        // B-operand: lane (<16) = key column ns*16+l16, kd contiguous in d
        const __bf16* kp = &sK[(ns * 16 + l16) * KSTR + c * 32 + (hi ? 16 : 0)];
        v16bf bbv = ld_b16x16(kp);
        sacc[ns] = __builtin_amdgcn_wmma_f32_16x16x32_bf16(
            false, qa[c], false, bbv, (short)0, sacc[ns], false, false);
      }
    }

    // ---- scale + causal mask (reference fills -1e9 post-scale) ----
#pragma unroll
    for (int ns = 0; ns < 2; ++ns) {
      const int j = kvbase + ns * 16 + l16;
#pragma unroll
      for (int v = 0; v < 8; ++v) {
        float s = sacc[ns][v] * scale;
        const int i = qw + v + rowoff;
        if (need_mask && j > i) s = -1.0e9f;
        sacc[ns][v] = s;
      }
    }

    // ---- online softmax: per-row reductions across 16-lane halves ----
    float alpha[8];
#pragma unroll
    for (int v = 0; v < 8; ++v) {
      float tm = fmaxf(sacc[0][v], sacc[1][v]);
      tm = fmaxf(tm, __shfl_xor(tm, 1, 32));
      tm = fmaxf(tm, __shfl_xor(tm, 2, 32));
      tm = fmaxf(tm, __shfl_xor(tm, 4, 32));
      tm = fmaxf(tm, __shfl_xor(tm, 8, 32));
      const float mnew = fmaxf(mrow[v], tm);
      const float a  = __expf(mrow[v] - mnew);
      const float p0 = __expf(sacc[0][v] - mnew);
      const float p1 = __expf(sacc[1][v] - mnew);
      float rs = p0 + p1;
      rs += __shfl_xor(rs, 1, 32);
      rs += __shfl_xor(rs, 2, 32);
      rs += __shfl_xor(rs, 4, 32);
      rs += __shfl_xor(rs, 8, 32);
      lrow[v] = lrow[v] * a + rs;
      mrow[v] = mnew;
      alpha[v] = a;
      sacc[0][v] = p0;
      sacc[1][v] = p1;
    }
#pragma unroll
    for (int t = 0; t < 8; ++t)
#pragma unroll
      for (int v = 0; v < 8; ++v) oacc[t][v] *= alpha[v];

    // ---- reshape P: C-layout regs -> LDS -> A-layout regs (wave-private scratch) ----
#pragma unroll
    for (int ns = 0; ns < 2; ++ns)
#pragma unroll
      for (int v = 0; v < 8; ++v)
        Pmy[(v + rowoff) * PSTR + ns * 16 + l16] = f2bf(sacc[ns][v]);
    // compiler inserts s_wait_dscnt before the dependent reads below
    const __bf16* pp = &Pmy[l16 * PSTR + 2 * kb];
    v16bf pav;
    {
      v8bf a0 = *(const v8bf*)pp;          // K = 2kb .. 2kb+7
      v8bf a1 = *(const v8bf*)(pp + 8);    // K = 2kb+8 .. 2kb+15
#pragma unroll
      for (int j = 0; j < 8; ++j) { pav[j] = a0[j]; pav[8 + j] = a1[j]; }
    }

    // ---- O += P V : 8 output tiles across dv, Kdim = 32 keys ----
#pragma unroll
    for (int t = 0; t < 8; ++t) {
      const __bf16* vp = &sVT[(t * 16 + l16) * VTSTR + (hi ? 16 : 0)];
      v16bf vbv = ld_b16x16(vp);
      oacc[t] = __builtin_amdgcn_wmma_f32_16x16x32_bf16(
          false, pav, false, vbv, (short)0, oacc[t], false, false);
    }
  }

  // ---- epilogue: O / l ----
  float linv[8];
#pragma unroll
  for (int v = 0; v < 8; ++v) linv[v] = 1.0f / lrow[v];
#pragma unroll
  for (int t = 0; t < 8; ++t)
#pragma unroll
    for (int v = 0; v < 8; ++v)
      Ob[(size_t)(qw + v + rowoff) * DIM + t * 16 + l16] = oacc[t][v] * linv[v];
}

extern "C" void kernel_launch(void* const* d_in, const int* in_sizes, int n_in,
                              void* d_out, int out_size, void* d_ws, size_t ws_size,
                              hipStream_t stream) {
  (void)in_sizes; (void)n_in; (void)out_size; (void)d_ws; (void)ws_size;
  const float* Q = (const float*)d_in[0];
  const float* K = (const float*)d_in[1];
  const float* V = (const float*)d_in[2];
  float* O = (float*)d_out;
  dim3 grid(SEQ / BM, BATCH);
  MaskedScaledDotProductAttention_67585605370702_kernel<<<grid, 256, 0, stream>>>(Q, K, V, O);
}
